// M11_brevitas_30855045054533
// MI455X (gfx1250) — 8-bit matrix operand order repaired and checked by reading, compiled, not run on hardware
//
#include <hip/hip_runtime.h>
#include <stdint.h>

typedef __attribute__((ext_vector_type(8))) int v8i;

__global__ void k_prep_sign(const float* __restrict__ w, int8_t* __restrict__ ws,
                            int Cout, int Cin, int K) {
  int idx = blockIdx.x * blockDim.x + threadIdx.x;
  int tot = Cout * Cin * K;
  if (idx >= tot) return;
  int k  = idx % K;
  int ci = (idx / K) % Cin;
  int co = idx / (K * Cin);
  ws[((size_t)co * K + k) * Cin + ci] = (w[idx] >= 0.0f) ? (int8_t)1 : (int8_t)-1;
}

__global__ void k_prep_conv1w(const float* __restrict__ w, float* __restrict__ sw, int n) {
  int i = blockIdx.x * blockDim.x + threadIdx.x;
  if (i < n) sw[i] = (w[i] >= 0.0f) ? 0.1f : -0.1f;
}

__global__ void k_conv1_fused(const float* __restrict__ x, const float* __restrict__ sw,
                              const float* __restrict__ cb, const float* __restrict__ g,
                              const float* __restrict__ bb, const float* __restrict__ bm,
                              const float* __restrict__ bv, int8_t* __restrict__ out) {
  const int C = 64, Lp = 995, Lin = 16000;
  int idx = blockIdx.x * blockDim.x + threadIdx.x;
  if (idx >= 256 * Lp * C) return;
  int oc = idx % C;
  int p  = (idx / C) % Lp;
  int n  = idx / (C * Lp);
  const float* xr = x + (size_t)n * Lin + 16 * p;
  const float* wr = sw + oc * 80;
  float a0 = 0.f, a1 = 0.f, a2 = 0.f, a3 = 0.f;
#pragma unroll 8
  for (int k = 0; k < 80; ++k) {
    float wk = wr[k];
    a0 = fmaf(xr[k],      wk, a0);
    a1 = fmaf(xr[k + 4],  wk, a1);
    a2 = fmaf(xr[k + 8],  wk, a2);
    a3 = fmaf(xr[k + 12], wk, a3);
  }
  float s   = g[oc] * rsqrtf(bv[oc] + 1e-5f);
  float off = cb[oc] - bm[oc];
  float bo  = bb[oc];
  bool any = ((a0 + off) * s + bo >= 0.0f) || ((a1 + off) * s + bo >= 0.0f) ||
             ((a2 + off) * s + bo >= 0.0f) || ((a3 + off) * s + bo >= 0.0f);
  out[idx] = any ? (int8_t)1 : (int8_t)-1;
}

template <int OUT_FLOAT>
__global__ void k_bconv_wmma(const int8_t* __restrict__ in, const int8_t* __restrict__ wsign,
                             const float* __restrict__ cb, const float* __restrict__ g,
                             const float* __restrict__ bb, const float* __restrict__ bm,
                             const float* __restrict__ bv,
                             int8_t* __restrict__ outq, float* __restrict__ outf,
                             int N, int Lin, int Cin, int Lout, int Cout, int Ktaps) {
  const int Ktot  = Ktaps * Cin;
  const int Ntot  = N * Lout;
  const int numM  = Cout >> 4;
  const int numNB = (Ntot + 63) >> 6;
  int wid = (blockIdx.x * blockDim.x + threadIdx.x) >> 5;
  if (wid >= numM * numNB) return;
  int mt   = wid % numM;
  int nb   = wid / numM;
  int lane = threadIdx.x & 31;
  int r    = lane & 15;
  int half = lane >> 4;

  const int8_t* arow = wsign + (size_t)(mt * 16 + r) * Ktot;

  const int8_t* bcol[4];
  int pos[4], nn[4], ll[4];
#pragma unroll
  for (int j = 0; j < 4; ++j) {
    pos[j] = nb * 64 + j * 16 + r;
    int cp = pos[j] < Ntot ? pos[j] : Ntot - 1;
    nn[j]  = cp / Lout;
    ll[j]  = cp - nn[j] * Lout;
    bcol[j] = in + ((size_t)nn[j] * Lin + ll[j]) * (size_t)Cin;
  }

  v8i acc[4] = {};
  for (int kc = 0; kc < Ktot; kc += 64) {
    const int2* ap = (const int2*)(arow + kc);
    int2 A0 = ap[2 * half], A1 = ap[2 * half + 1], A2 = ap[4 + 2 * half], A3 = ap[5 + 2 * half];
    v8i a;
    a[0] = A0.x; a[1] = A0.y; a[2] = A1.x; a[3] = A1.y;
    a[4] = A2.x; a[5] = A2.y; a[6] = A3.x; a[7] = A3.y;
#pragma unroll
    for (int j = 0; j < 4; ++j) {
      const int4* bp = (const int4*)(bcol[j] + kc);
      int4 B0 = bp[half], B1 = bp[2 + half];
      v8i b;
      b[0] = B0.x; b[1] = B0.y; b[2] = B0.z; b[3] = B0.w;
      b[4] = B1.x; b[5] = B1.y; b[6] = B1.z; b[7] = B1.w;
      acc[j] = __builtin_amdgcn_wmma_i32_16x16x64_iu8(true, a, true, b, acc[j], false, false);
    }
  }

  int chBase = mt * 16 + half * 8;
#pragma unroll
  for (int j = 0; j < 4; ++j) {
    if (pos[j] >= Ntot) continue;
    if (OUT_FLOAT) {
      float* dst = outf + (size_t)pos[j] * Cout + chBase;
#pragma unroll
      for (int rr = 0; rr < 8; ++rr) dst[rr] = 0.1f * (float)acc[j][rr] + cb[chBase + rr];
    } else {
      uint32_t lo = 0, hi = 0;
#pragma unroll
      for (int rr = 0; rr < 8; ++rr) {
        int ch  = chBase + rr;
        float s = g[ch] * rsqrtf(bv[ch] + 1e-5f);
        float t = (0.1f * (float)acc[j][rr] + cb[ch] - bm[ch]) * s + bb[ch];
        uint32_t q = (t >= 0.0f) ? 0x01u : 0xFFu;
        if (rr < 4) lo |= q << (8 * rr); else hi |= q << (8 * (rr - 4));
      }
      uint32_t* dst = (uint32_t*)(outq + ((size_t)nn[j] * Lout + ll[j]) * Cout + chBase);
      dst[0] = lo; dst[1] = hi;
    }
  }
}

__global__ void k_pool4(const int8_t* __restrict__ in, int8_t* __restrict__ out,
                        int N, int Lq, int Lp, int C) {
  int idx = blockIdx.x * blockDim.x + threadIdx.x;
  if (idx >= N * Lp * C) return;
  int c = idx % C;
  int p = (idx / C) % Lp;
  int n = idx / (C * Lp);
  const int8_t* s = in + ((size_t)n * Lq + 4 * p) * C + c;
  int8_t m = s[0];
  if (s[C]     > m) m = s[C];
  if (s[2 * C] > m) m = s[2 * C];
  if (s[3 * C] > m) m = s[3 * C];
  out[idx] = m;
}

__global__ void k_fc1_fused(const float* __restrict__ emb, const float* __restrict__ w,
                            const float* __restrict__ g, const float* __restrict__ bb,
                            const float* __restrict__ bm, const float* __restrict__ bv,
                            int8_t* __restrict__ out) {
  int idx = blockIdx.x * blockDim.x + threadIdx.x;
  if (idx >= 256 * 128) return;
  int i = idx % 128;
  int n = idx / 128;
  const float* xr = emb + (size_t)n * 512;
  const float* wr = w + (size_t)i * 512;
  float acc = 0.0f;
#pragma unroll 8
  for (int j = 0; j < 512; ++j) acc = fmaf(xr[j], (wr[j] >= 0.0f ? 0.1f : -0.1f), acc);
  float s = g[i] * rsqrtf(bv[i] + 1e-5f);
  float t = (acc - bm[i]) * s + bb[i];
  out[idx] = (t >= 0.0f) ? (int8_t)1 : (int8_t)-1;
}

__global__ void k_fc2(const int8_t* __restrict__ a, const float* __restrict__ w,
                      float* __restrict__ out) {
  int idx = blockIdx.x * blockDim.x + threadIdx.x;
  if (idx >= 256 * 35) return;
  int o = idx % 35;
  int n = idx / 35;
  const int8_t* ar = a + (size_t)n * 128;
  const float*  wr = w + (size_t)o * 128;
  int acc = 0;
#pragma unroll 8
  for (int j = 0; j < 128; ++j) acc += (int)ar[j] * (wr[j] >= 0.0f ? 1 : -1);
  out[idx] = 0.1f * (float)acc;
}

extern "C" void kernel_launch(void* const* d_in, const int* in_sizes, int n_in,
                              void* d_out, int out_size, void* d_ws, size_t ws_size,
                              hipStream_t stream) {
  (void)in_sizes; (void)n_in; (void)out_size; (void)ws_size;
  const float* x = (const float*)d_in[0];
  const float* cw[12]; const float* cbias[12];
  for (int i = 0; i < 12; ++i) { cw[i] = (const float*)d_in[1 + 2 * i]; cbias[i] = (const float*)d_in[2 + 2 * i]; }
  const float *bg[12], *bB[12], *bM[12], *bV[12];
  for (int i = 0; i < 12; ++i) {
    bg[i] = (const float*)d_in[25 + 4 * i];
    bB[i] = (const float*)d_in[26 + 4 * i];
    bM[i] = (const float*)d_in[27 + 4 * i];
    bV[i] = (const float*)d_in[28 + 4 * i];
  }
  const float* fc1w = (const float*)d_in[73];
  const float* fc2w = (const float*)d_in[74];

  uint8_t* ws = (uint8_t*)d_ws;
  size_t off = 0;
  auto alloc = [&](size_t bytes) -> uint8_t* {
    uint8_t* p = ws + off; off = (off + bytes + 255) & ~(size_t)255; return p;
  };
  const size_t ACT = (size_t)256 * 995 * 64 + 4096;
  int8_t* bufA = (int8_t*)alloc(ACT);
  int8_t* bufB = (int8_t*)alloc(ACT);
  int8_t* bufC = (int8_t*)alloc(ACT);
  float*  sw1  = (float*)alloc((size_t)64 * 80 * 4);
  static const int Cout_[12] = {64, 64, 64, 128, 128, 256, 256, 256, 256, 512, 512, 512};
  static const int Cin_[12]  = {1, 64, 64, 64, 128, 128, 256, 256, 256, 256, 512, 512};
  static const int Kt_[12]   = {80, 3, 3, 3, 3, 3, 3, 3, 3, 3, 3, 2};
  int8_t* wsg[12];
  for (int i = 1; i < 12; ++i) wsg[i] = (int8_t*)alloc((size_t)Cout_[i] * Cin_[i] * Kt_[i]);
  float*  embout = (float*)alloc((size_t)256 * 512 * 4);
  int8_t* fc1a   = (int8_t*)alloc((size_t)256 * 128);

  k_prep_conv1w<<<(64 * 80 + 255) / 256, 256, 0, stream>>>(cw[0], sw1, 64 * 80);
  for (int i = 1; i < 12; ++i) {
    int tot = Cout_[i] * Cin_[i] * Kt_[i];
    k_prep_sign<<<(tot + 255) / 256, 256, 0, stream>>>(cw[i], wsg[i], Cout_[i], Cin_[i], Kt_[i]);
  }

  {
    int tot = 256 * 995 * 64;
    k_conv1_fused<<<(tot + 255) / 256, 256, 0, stream>>>(x, sw1, cbias[0], bg[0], bB[0], bM[0], bV[0], bufA);
  }

  auto bconv = [&](int ci, const int8_t* in, int Lin, int8_t* outq, float* outf) -> int {
    int Cin = Cin_[ci], Cout = Cout_[ci], K = Kt_[ci];
    int Lout = Lin - K + 1;
    int tiles = (Cout / 16) * ((256 * Lout + 63) / 64);
    int blocks = (tiles + 7) / 8;
    if (outf)
      k_bconv_wmma<1><<<blocks, 256, 0, stream>>>(in, wsg[ci], cbias[ci], bg[ci], bB[ci], bM[ci], bV[ci],
                                                  nullptr, outf, 256, Lin, Cin, Lout, Cout, K);
    else
      k_bconv_wmma<0><<<blocks, 256, 0, stream>>>(in, wsg[ci], cbias[ci], bg[ci], bB[ci], bM[ci], bV[ci],
                                                  outq, nullptr, 256, Lin, Cin, Lout, Cout, K);
    return Lout;
  };
  auto pool = [&](const int8_t* in, int8_t* out, int Lq, int C) -> int {
    int Lp = Lq / 4;
    int tot = 256 * Lp * C;
    k_pool4<<<(tot + 255) / 256, 256, 0, stream>>>(in, out, 256, Lq, Lp, C);
    return Lp;
  };

  int L = 995;
  L = bconv(1,  bufA, L, bufC, nullptr);
  L = bconv(2,  bufC, L, bufA, nullptr);
  L = pool(bufA, bufB, L, 64);
  L = bconv(3,  bufB, L, bufA, nullptr);
  L = bconv(4,  bufA, L, bufC, nullptr);
  L = pool(bufC, bufB, L, 128);
  L = bconv(5,  bufB, L, bufA, nullptr);
  L = bconv(6,  bufA, L, bufC, nullptr);
  L = bconv(7,  bufC, L, bufA, nullptr);
  L = bconv(8,  bufA, L, bufC, nullptr);
  L = pool(bufC, bufB, L, 256);
  L = bconv(9,  bufB, L, bufA, nullptr);
  L = bconv(10, bufA, L, bufC, nullptr);
  L = pool(bufC, bufB, L, 512);
  L = bconv(11, bufB, L, nullptr, embout);

  k_fc1_fused<<<(256 * 128 + 255) / 256, 256, 0, stream>>>(embout, fc1w, bg[11], bB[11], bM[11], bV[11], fc1a);
  k_fc2<<<(256 * 35 + 255) / 256, 256, 0, stream>>>(fc1a, fc2w, (float*)d_out);
}
